// AttentionHead_22290880266281
// MI455X (gfx1250) — matrix operand order repaired and checked by reading, not compiled, not run on hardware
//
#include <hip/hip_runtime.h>
#include <math.h>

#define E_DIM 1024
#define B_DIM 8
#define T_DIM 2048
#define M_DIM (B_DIM * T_DIM)   // 16384 total rows across batch
#define LD_T 40                  // LDS row stride in bf16 for 32-wide tiles (80B, 16B-aligned)

typedef __bf16 bf16_t;
typedef __attribute__((ext_vector_type(8)))  __bf16 v8bf;
typedef __attribute__((ext_vector_type(16))) __bf16 v16bf;
typedef __attribute__((ext_vector_type(8)))  float  v8f;

#define NEG_INF (-__builtin_inff())

__device__ __forceinline__ v16bf cat16(v8bf lo, v8bf hi) {
  v16bf r;
#pragma unroll
  for (int i = 0; i < 8; ++i) { r[i] = lo[i]; r[i + 8] = hi[i]; }
  return r;
}

// A fragment (16x32 bf16, ISA 7.12.2): lane&15 = row M; lanes 0-15 hold K={0..7,16..23},
// lanes 16-31 hold K={8..15,24..31}. tile points at subtile origin, row stride LD_T.
__device__ __forceinline__ v16bf load_a_frag(const bf16_t* tile, int lane) {
  int r = lane & 15, h = lane >> 4;
  const bf16_t* p = tile + r * LD_T;
  v8bf lo = *(const v8bf*)(p + h * 8);
  v8bf hi = *(const v8bf*)(p + 16 + h * 8);
  return cat16(lo, hi);
}

// B fragment (32x16 bf16): lane&15 = column N; lanes 0-15 hold K={0..7,16..23}, lanes 16-31 K={8..15,24..31}.
// tile layout in LDS is [column][K] with row stride LD_T -> two 16B reads per lane.
__device__ __forceinline__ v16bf load_b_frag(const bf16_t* tile, int lane) {
  int c = lane & 15, h = lane >> 4;
  const bf16_t* p = tile + c * LD_T;
  return cat16(*(const v8bf*)(p + h * 8), *(const v8bf*)(p + 16 + h * 8));
}

__device__ __forceinline__ v8f wmma_bf16(v16bf a, v16bf b, v8f c) {
  // (neg_a, A, neg_b, B, c_mod, C, reuse_a, reuse_b)
  return __builtin_amdgcn_wmma_f32_16x16x32_bf16(false, a, false, b, (short)0, c, false, false);
}

// ---------------------------------------------------------------------------
// K1: fp32 weights -> bf16
// ---------------------------------------------------------------------------
__global__ __launch_bounds__(256) void wconv_kernel(const float* __restrict__ src,
                                                    bf16_t* __restrict__ dst, int n) {
  int i = (blockIdx.x * 256 + threadIdx.x) * 4;
  if (i + 3 < n) {
    float4 f = *(const float4*)(src + i);
    dst[i + 0] = (bf16_t)f.x; dst[i + 1] = (bf16_t)f.y;
    dst[i + 2] = (bf16_t)f.z; dst[i + 3] = (bf16_t)f.w;
  }
}

// ---------------------------------------------------------------------------
// K2: Q/K/V = X * W^T + b   (blockIdx.z selects the projection)
// WG tile 256(M) x 64(N); wave owns 32x64 -> acc[2][4], 8 WMMA per 12 b128 LDS
// loads. Double-buffered LDS so global fetch of tile k+1 overlaps WMMA of k.
// ---------------------------------------------------------------------------
__global__ __launch_bounds__(256) void qkv_gemm_kernel(const float* __restrict__ X,
    const bf16_t* __restrict__ W3, const float* __restrict__ bq,
    const float* __restrict__ bk, const float* __restrict__ bv,
    bf16_t* __restrict__ QKV) {
  __shared__ bf16_t As[2][256 * LD_T];
  __shared__ bf16_t Bs[2][64 * LD_T];
  const int z = blockIdx.z;
  const bf16_t* W = W3 + (size_t)z * E_DIM * E_DIM;
  const float* bias = (z == 0) ? bq : (z == 1 ? bk : bv);
  bf16_t* Out = QKV + (size_t)z * M_DIM * E_DIM;
  const int m0 = blockIdx.y * 256;
  const int n0 = blockIdx.x * 64;
  const int tid = threadIdx.x;
  const int w = tid >> 5, lane = tid & 31, half = lane >> 4, c = lane & 15;

  v8f acc[2][4];
#pragma unroll
  for (int mi = 0; mi < 2; ++mi)
#pragma unroll
    for (int s = 0; s < 4; ++s)
#pragma unroll
      for (int v = 0; v < 8; ++v) acc[mi][s][v] = 0.f;

  auto stage = [&](int buf, int k0) {
    // A: X[m0..m0+256)[k0..k0+32) fp32 -> bf16 LDS (8 float4 per thread)
#pragma unroll
    for (int it = 0; it < 8; ++it) {
      int idx = tid + it * 256;
      int row = idx >> 3, c4 = idx & 7;
      float4 f = *(const float4*)(X + (size_t)(m0 + row) * E_DIM + k0 + c4 * 4);
      bf16_t* d = As[buf] + row * LD_T + c4 * 4;
      d[0] = (bf16_t)f.x; d[1] = (bf16_t)f.y; d[2] = (bf16_t)f.z; d[3] = (bf16_t)f.w;
    }
    // B: W[n0..n0+64)[k0..k0+32) (contraction dim contiguous -> no transpose)
    int row = tid >> 2, c8 = tid & 3;
    *(v8bf*)(Bs[buf] + row * LD_T + c8 * 8) =
        *(const v8bf*)(W + (size_t)(n0 + row) * E_DIM + k0 + c8 * 8);
  };

  stage(0, 0);
  __syncthreads();
  for (int k0 = 0; k0 < E_DIM; k0 += 32) {
    const int cur = (k0 >> 5) & 1;
    if (k0 + 32 < E_DIM) stage(cur ^ 1, k0 + 32);
    v16bf a0 = load_a_frag(As[cur] + (32 * w) * LD_T, lane);
    v16bf a1 = load_a_frag(As[cur] + (32 * w + 16) * LD_T, lane);
#pragma unroll
    for (int s = 0; s < 4; ++s) {
      v16bf b = load_b_frag(Bs[cur] + (16 * s) * LD_T, lane);
      acc[0][s] = wmma_bf16(a0, b, acc[0][s]);
      acc[1][s] = wmma_bf16(a1, b, acc[1][s]);
    }
    __syncthreads();
  }
#pragma unroll
  for (int s = 0; s < 4; ++s) {
    int col = n0 + 16 * s + c;
    float bcol = bias[col];
#pragma unroll
    for (int mi = 0; mi < 2; ++mi)
#pragma unroll
      for (int v = 0; v < 8; ++v) {
        int row = m0 + 32 * w + 16 * mi + v + 8 * half;
        Out[(size_t)row * E_DIM + col] = (bf16_t)(acc[mi][s][v] + bcol);
      }
  }
}

// ---------------------------------------------------------------------------
// K3: per-column (query-axis) softmax stats. WG owns 64 columns of one batch;
// streams 256-row blocks i >= j with online max/sum (S via WMMA Q*K^T).
// ---------------------------------------------------------------------------
__global__ __launch_bounds__(256) void colstats_kernel(const bf16_t* __restrict__ QKV,
    float* __restrict__ colmax, float* __restrict__ colsum) {
  __shared__ bf16_t Qs[2][256 * LD_T];
  __shared__ bf16_t Ks[2][64 * LD_T];
  __shared__ float redm[8][64];
  __shared__ float redl[8][64];
  const int b = blockIdx.y;
  const int j0 = blockIdx.x * 64;
  const int tid = threadIdx.x, w = tid >> 5, lane = tid & 31, half = lane >> 4, c = lane & 15;
  const bf16_t* Q  = QKV + (size_t)b * T_DIM * E_DIM;
  const bf16_t* Km = QKV + (size_t)M_DIM * E_DIM + (size_t)b * T_DIM * E_DIM;
  const float sc = 0.03125f;  // 1/sqrt(1024)

  float mrun[4], lrun[4];
#pragma unroll
  for (int s = 0; s < 4; ++s) { mrun[s] = NEG_INF; lrun[s] = 0.f; }

  for (int i0 = (j0 / 256) * 256; i0 < T_DIM; i0 += 256) {
    v8f acc[2][4];
#pragma unroll
    for (int mi = 0; mi < 2; ++mi)
#pragma unroll
      for (int s = 0; s < 4; ++s)
#pragma unroll
        for (int v = 0; v < 8; ++v) acc[mi][s][v] = 0.f;

    auto stage = [&](int buf, int e0) {
#pragma unroll
      for (int it = 0; it < 4; ++it) {
        int idx = tid + it * 256;
        int row = idx >> 2, c8 = idx & 3;
        *(v8bf*)(Qs[buf] + row * LD_T + c8 * 8) =
            *(const v8bf*)(Q + (size_t)(i0 + row) * E_DIM + e0 + c8 * 8);
      }
      int row = tid >> 2, c8 = tid & 3;
      *(v8bf*)(Ks[buf] + row * LD_T + c8 * 8) =
          *(const v8bf*)(Km + (size_t)(j0 + row) * E_DIM + e0 + c8 * 8);
    };

    stage(0, 0);
    __syncthreads();
    for (int e0 = 0; e0 < E_DIM; e0 += 32) {
      const int cur = (e0 >> 5) & 1;
      if (e0 + 32 < E_DIM) stage(cur ^ 1, e0 + 32);
      v16bf a0 = load_a_frag(Qs[cur] + (32 * w) * LD_T, lane);
      v16bf a1 = load_a_frag(Qs[cur] + (32 * w + 16) * LD_T, lane);
#pragma unroll
      for (int s = 0; s < 4; ++s) {
        v16bf bfr = load_b_frag(Ks[cur] + (16 * s) * LD_T, lane);
        acc[0][s] = wmma_bf16(a0, bfr, acc[0][s]);
        acc[1][s] = wmma_bf16(a1, bfr, acc[1][s]);
      }
      __syncthreads();
    }
    // online per-column update; a lane owns one column (duplicated in lane^16).
    // Both M-subtiles of this wave feed the same per-column stats.
#pragma unroll
    for (int s = 0; s < 4; ++s) {
      int col = j0 + 16 * s + c;
      float sv[2][8], tmax = NEG_INF;
#pragma unroll
      for (int mi = 0; mi < 2; ++mi)
#pragma unroll
        for (int v = 0; v < 8; ++v) {
          int row = i0 + 32 * w + 16 * mi + v + 8 * half;
          float x = acc[mi][s][v] * sc;
          if (col > row) x = NEG_INF;   // causal mask
          sv[mi][v] = x;
          tmax = fmaxf(tmax, x);
        }
      tmax = fmaxf(tmax, __shfl_xor(tmax, 16, 32));
      float mnew = fmaxf(mrun[s], tmax);
      if (mnew != NEG_INF) {
        float psum = 0.f;
#pragma unroll
        for (int mi = 0; mi < 2; ++mi)
#pragma unroll
          for (int v = 0; v < 8; ++v) psum += __expf(sv[mi][v] - mnew);
        psum += __shfl_xor(psum, 16, 32);
        float fct = (mrun[s] == NEG_INF) ? 0.f : __expf(mrun[s] - mnew);
        lrun[s] = lrun[s] * fct + psum;
        mrun[s] = mnew;
      }
    }
  }
  if (half == 0) {
#pragma unroll
    for (int s = 0; s < 4; ++s) { redm[w][16 * s + c] = mrun[s]; redl[w][16 * s + c] = lrun[s]; }
  }
  __syncthreads();
  if (tid < 64) {
    float M_ = NEG_INF;
#pragma unroll
    for (int ww = 0; ww < 8; ++ww) M_ = fmaxf(M_, redm[ww][tid]);
    float L = 0.f;
#pragma unroll
    for (int ww = 0; ww < 8; ++ww) {
      float mm = redm[ww][tid];
      if (mm != NEG_INF) L += redl[ww][tid] * __expf(mm - M_);
    }
    colmax[(size_t)b * T_DIM + j0 + tid] = M_;
    colsum[(size_t)b * T_DIM + j0 + tid] = L;
  }
}

// ---------------------------------------------------------------------------
// K4: materialize P = exp(S - colmax)/colsum (bf16), masked entries = 0.
// ---------------------------------------------------------------------------
__global__ __launch_bounds__(256) void pmat_kernel(const bf16_t* __restrict__ QKV,
    const float* __restrict__ colmax, const float* __restrict__ colsum,
    bf16_t* __restrict__ P) {
  const int j0 = blockIdx.x * 64;
  const int i0 = blockIdx.y * 256;
  if (j0 > i0 + 255) return;  // fully above diagonal: never read downstream
  __shared__ bf16_t Qs[2][256 * LD_T];
  __shared__ bf16_t Ks[2][64 * LD_T];
  const int b = blockIdx.z;
  const int tid = threadIdx.x, w = tid >> 5, lane = tid & 31, half = lane >> 4, c = lane & 15;
  const bf16_t* Q  = QKV + (size_t)b * T_DIM * E_DIM;
  const bf16_t* Km = QKV + (size_t)M_DIM * E_DIM + (size_t)b * T_DIM * E_DIM;
  const float sc = 0.03125f;

  v8f acc[2][4];
#pragma unroll
  for (int mi = 0; mi < 2; ++mi)
#pragma unroll
    for (int s = 0; s < 4; ++s)
#pragma unroll
      for (int v = 0; v < 8; ++v) acc[mi][s][v] = 0.f;

  auto stage = [&](int buf, int e0) {
#pragma unroll
    for (int it = 0; it < 4; ++it) {
      int idx = tid + it * 256;
      int row = idx >> 2, c8 = idx & 3;
      *(v8bf*)(Qs[buf] + row * LD_T + c8 * 8) =
          *(const v8bf*)(Q + (size_t)(i0 + row) * E_DIM + e0 + c8 * 8);
    }
    int row = tid >> 2, c8 = tid & 3;
    *(v8bf*)(Ks[buf] + row * LD_T + c8 * 8) =
        *(const v8bf*)(Km + (size_t)(j0 + row) * E_DIM + e0 + c8 * 8);
  };

  stage(0, 0);
  __syncthreads();
  for (int e0 = 0; e0 < E_DIM; e0 += 32) {
    const int cur = (e0 >> 5) & 1;
    if (e0 + 32 < E_DIM) stage(cur ^ 1, e0 + 32);
    v16bf a0 = load_a_frag(Qs[cur] + (32 * w) * LD_T, lane);
    v16bf a1 = load_a_frag(Qs[cur] + (32 * w + 16) * LD_T, lane);
#pragma unroll
    for (int s = 0; s < 4; ++s) {
      v16bf bfr = load_b_frag(Ks[cur] + (16 * s) * LD_T, lane);
      acc[0][s] = wmma_bf16(a0, bfr, acc[0][s]);
      acc[1][s] = wmma_bf16(a1, bfr, acc[1][s]);
    }
    __syncthreads();
  }
  bf16_t* Pb = P + (size_t)b * T_DIM * T_DIM;
#pragma unroll
  for (int s = 0; s < 4; ++s) {
    int col = j0 + 16 * s + c;
    float cm  = colmax[(size_t)b * T_DIM + col];
    float rcs = 1.0f / colsum[(size_t)b * T_DIM + col];
#pragma unroll
    for (int mi = 0; mi < 2; ++mi)
#pragma unroll
      for (int v = 0; v < 8; ++v) {
        int row = i0 + 32 * w + 16 * mi + v + 8 * half;
        float x = acc[mi][s][v] * sc;
        float p = (col > row) ? 0.f : __expf(x - cm) * rcs;
        Pb[(size_t)row * T_DIM + col] = (bf16_t)p;
      }
  }
}

// ---------------------------------------------------------------------------
// K5: O = P * V (fp32 out). Contraction dim j is V's row index, so V tiles are
// transposed through LDS to keep B-fragment loads contiguous.
// ---------------------------------------------------------------------------
__global__ __launch_bounds__(256) void out_gemm_kernel(const bf16_t* __restrict__ P,
    const bf16_t* __restrict__ QKV, float* __restrict__ Out) {
  __shared__ bf16_t As[2][256 * LD_T];
  __shared__ bf16_t Vt[2][64 * LD_T];
  const int b = blockIdx.z;
  const int n0 = blockIdx.x * 64;
  const int m0 = blockIdx.y * 256;
  const int tid = threadIdx.x, w = tid >> 5, lane = tid & 31, half = lane >> 4, c = lane & 15;
  const bf16_t* Pb = P + (size_t)b * T_DIM * T_DIM;
  const bf16_t* V  = QKV + 2 * (size_t)M_DIM * E_DIM + (size_t)b * T_DIM * E_DIM;

  v8f acc[2][4];
#pragma unroll
  for (int mi = 0; mi < 2; ++mi)
#pragma unroll
    for (int s = 0; s < 4; ++s)
#pragma unroll
      for (int v = 0; v < 8; ++v) acc[mi][s][v] = 0.f;

  auto stage = [&](int buf, int j0) {
#pragma unroll
    for (int it = 0; it < 4; ++it) {
      int idx = tid + it * 256;
      int row = idx >> 2, c8 = idx & 3;
      *(v8bf*)(As[buf] + row * LD_T + c8 * 8) =
          *(const v8bf*)(Pb + (size_t)(m0 + row) * T_DIM + j0 + c8 * 8);
    }
    int j = tid >> 3, c8 = tid & 7;
    v8bf val = *(const v8bf*)(V + (size_t)(j0 + j) * E_DIM + n0 + c8 * 8);
#pragma unroll
    for (int u = 0; u < 8; ++u) Vt[buf][(c8 * 8 + u) * LD_T + j] = val[u];
  };

  const int jend = m0 + 256;  // P is zero (or unread) beyond the diagonal block
  stage(0, 0);
  __syncthreads();
  for (int j0 = 0; j0 < jend; j0 += 32) {
    const int cur = (j0 >> 5) & 1;
    if (j0 + 32 < jend) stage(cur ^ 1, j0 + 32);
    v16bf a0 = load_a_frag(As[cur] + (32 * w) * LD_T, lane);
    v16bf a1 = load_a_frag(As[cur] + (32 * w + 16) * LD_T, lane);
#pragma unroll
    for (int s = 0; s < 4; ++s) {
      v16bf bfr = load_b_frag(Vt[cur] + (16 * s) * LD_T, lane);
      acc[0][s] = wmma_bf16(a0, bfr, acc[0][s]);
      acc[1][s] = wmma_bf16(a1, bfr, acc[1][s]);
    }
    __syncthreads();
  }
#pragma unroll
  for (int s = 0; s < 4; ++s) {
    int col = n0 + 16 * s + c;
#pragma unroll
    for (int mi = 0; mi < 2; ++mi)
#pragma unroll
      for (int v = 0; v < 8; ++v) {
        int row = m0 + 32 * w + 16 * mi + v + 8 * half;
        Out[((size_t)b * T_DIM + row) * E_DIM + col] = acc[mi][s][v];
      }
  }
}

// ---------------------------------------------------------------------------
extern "C" void kernel_launch(void* const* d_in, const int* in_sizes, int n_in,
                              void* d_out, int out_size, void* d_ws, size_t ws_size,
                              hipStream_t stream) {
  (void)in_sizes; (void)n_in; (void)out_size; (void)ws_size;
  const float* X  = (const float*)d_in[0];
  const float* Wq = (const float*)d_in[1];
  const float* bq = (const float*)d_in[2];
  const float* Wk = (const float*)d_in[3];
  const float* bk = (const float*)d_in[4];
  const float* Wv = (const float*)d_in[5];
  const float* bv = (const float*)d_in[6];
  float* Out = (float*)d_out;

  char* ws = (char*)d_ws;
  size_t off = 0;
  auto alloc = [&](size_t bytes) -> char* {
    char* p = ws + off;
    off += (bytes + 255) & ~(size_t)255;
    return p;
  };
  bf16_t* w3     = (bf16_t*)alloc((size_t)3 * E_DIM * E_DIM * sizeof(bf16_t));      // 6 MB
  bf16_t* qkv    = (bf16_t*)alloc((size_t)3 * M_DIM * E_DIM * sizeof(bf16_t));      // 100 MB
  float*  colmax = (float*) alloc((size_t)B_DIM * T_DIM * sizeof(float));
  float*  colsum = (float*) alloc((size_t)B_DIM * T_DIM * sizeof(float));
  bf16_t* Pmat   = (bf16_t*)alloc((size_t)B_DIM * T_DIM * T_DIM * sizeof(bf16_t));  // 67 MB

  const int nW = E_DIM * E_DIM;
  wconv_kernel<<<dim3(nW / 1024), dim3(256), 0, stream>>>(Wq, w3 + 0 * (size_t)nW, nW);
  wconv_kernel<<<dim3(nW / 1024), dim3(256), 0, stream>>>(Wk, w3 + 1 * (size_t)nW, nW);
  wconv_kernel<<<dim3(nW / 1024), dim3(256), 0, stream>>>(Wv, w3 + 2 * (size_t)nW, nW);

  qkv_gemm_kernel<<<dim3(E_DIM / 64, M_DIM / 256, 3), dim3(256), 0, stream>>>(
      X, w3, bq, bk, bv, qkv);

  colstats_kernel<<<dim3(T_DIM / 64, B_DIM), dim3(256), 0, stream>>>(qkv, colmax, colsum);

  pmat_kernel<<<dim3(T_DIM / 64, T_DIM / 256, B_DIM), dim3(256), 0, stream>>>(
      qkv, colmax, colsum, Pmat);

  out_gemm_kernel<<<dim3(E_DIM / 64, T_DIM / 256, B_DIM), dim3(256), 0, stream>>>(
      Pmat, qkv, Out);
}
